// Decoder_Select_29703993819600
// MI455X (gfx1250) — matrix operand order repaired and checked by reading, not compiled, not run on hardware
//
#include <hip/hip_runtime.h>
#include <hip/hip_bf16.h>
#include <math.h>

typedef __attribute__((ext_vector_type(16))) __bf16 v16bf;
typedef __attribute__((ext_vector_type(8)))  __bf16 v8bf;
typedef __attribute__((ext_vector_type(8)))  float  v8f;

#define IN_CH   64
#define BNCH    128
#define CKSZ    100
#define KFR     127
#define NPIX    (CKSZ * KFR)   // 12700
#define NFRM    6200
#define TOUT    49608          // (6200-1)*8 + 16
#define NSAMP   8
#define LDA     136            // padded bf16 stride (272B: rotates banks, 16B aligned)
#define LDB     136

__device__ __forceinline__ unsigned short f2bfu(float x) {
  unsigned u = __builtin_bit_cast(unsigned, x);
  return (unsigned short)((u + 0x7FFFu + ((u >> 16) & 1u)) >> 16);   // RNE
}
__device__ __forceinline__ __bf16 f2bf(float x) {
  unsigned short h = f2bfu(x);
  return __builtin_bit_cast(__bf16, h);
}

#define WMMA_BF16(A, B, C) \
  __builtin_amdgcn_wmma_f32_16x16x32_bf16(false, (A), false, (B), (short)0, (C), false, false)

// ---- stage 32 rows x 128 K of fp32 weights into LDS as bf16 [row][k] (coalesced b128 loads)
__device__ __forceinline__ void stage_w(const float* __restrict__ W, int m0,
                                        __bf16* lA, int tid) {
#pragma unroll
  for (int base = 0; base < 4096; base += 2048) {
    int idx = base + tid * 8;
    int r = idx >> 7, k0 = idx & 127;
    const float4* s4 = (const float4*)(W + (size_t)(m0 + r) * BNCH + k0);
    float4 u = s4[0], v = s4[1];
    __bf16* dst = lA + r * LDA + k0;
    dst[0] = f2bf(u.x); dst[1] = f2bf(u.y); dst[2] = f2bf(u.z); dst[3] = f2bf(u.w);
    dst[4] = f2bf(v.x); dst[5] = f2bf(v.y); dst[6] = f2bf(v.z); dst[7] = f2bf(v.w);
  }
}

// ---- stage 128 K x 128 N of activations into LDS, K-transposed bf16 [col][k].
// Edge blocks clamp the column (WMMA cols map 1:1 to lanes; clamped cols are never stored).
template <bool PRELU>
__device__ __forceinline__ void stage_x(const float* __restrict__ X, int ldn, int n0b,
                                        int nmax, float alpha, __bf16* lB, int tid) {
  const bool full = (n0b + 128) <= nmax;
  const int cl = nmax - 1;
#pragma unroll
  for (int i = 0; i < 8; ++i) {
    int chunk = tid + i * 256;          // 0..2047
    int k = (chunk >> 5) << 1;          // even k
    int c4 = (chunk & 31) << 2;         // col 0,4,...,124
    float a0, a1, a2, a3, b0, b1, b2, b3;
    if (full) {
      float4 u = *(const float4*)(X + (size_t)k * ldn + n0b + c4);
      float4 v = *(const float4*)(X + (size_t)(k + 1) * ldn + n0b + c4);
      a0 = u.x; a1 = u.y; a2 = u.z; a3 = u.w;
      b0 = v.x; b1 = v.y; b2 = v.z; b3 = v.w;
    } else {
      const float* r0 = X + (size_t)k * ldn;
      const float* r1 = r0 + ldn;
      int c0 = n0b + c4,     cc0 = c0 > cl ? cl : c0;
      int c1 = n0b + c4 + 1, cc1 = c1 > cl ? cl : c1;
      int c2 = n0b + c4 + 2, cc2 = c2 > cl ? cl : c2;
      int c3 = n0b + c4 + 3, cc3 = c3 > cl ? cl : c3;
      a0 = r0[cc0]; a1 = r0[cc1]; a2 = r0[cc2]; a3 = r0[cc3];
      b0 = r1[cc0]; b1 = r1[cc1]; b2 = r1[cc2]; b3 = r1[cc3];
    }
    if (PRELU) {
      a0 = a0 >= 0.f ? a0 : alpha * a0;  a1 = a1 >= 0.f ? a1 : alpha * a1;
      a2 = a2 >= 0.f ? a2 : alpha * a2;  a3 = a3 >= 0.f ? a3 : alpha * a3;
      b0 = b0 >= 0.f ? b0 : alpha * b0;  b1 = b1 >= 0.f ? b1 : alpha * b1;
      b2 = b2 >= 0.f ? b2 : alpha * b2;  b3 = b3 >= 0.f ? b3 : alpha * b3;
    }
    unsigned* dst0 = (unsigned*)(lB + (size_t)(c4 + 0) * LDB + k);
    unsigned* dst1 = (unsigned*)(lB + (size_t)(c4 + 1) * LDB + k);
    unsigned* dst2 = (unsigned*)(lB + (size_t)(c4 + 2) * LDB + k);
    unsigned* dst3 = (unsigned*)(lB + (size_t)(c4 + 3) * LDB + k);
    *dst0 = (unsigned)f2bfu(a0) | ((unsigned)f2bfu(b0) << 16);
    *dst1 = (unsigned)f2bfu(a1) | ((unsigned)f2bfu(b1) << 16);
    *dst2 = (unsigned)f2bfu(a2) | ((unsigned)f2bfu(b2) << 16);
    *dst3 = (unsigned)f2bfu(a3) | ((unsigned)f2bfu(b3) << 16);
  }
}

// ---- fragment fetch from LDS (two ds_load_b128 each), per ISA 7.12.2 wave32 layouts
__device__ __forceinline__ v16bf frag_a_lds(const __bf16* lA, int lane, int kk) {
  const __bf16* p = lA + (lane & 15) * LDA + kk + ((lane >> 4) << 3);
  const v8bf* pv = (const v8bf*)p;
  v8bf lo = pv[0];   // K = kk+half8 + 0..7
  v8bf hi = pv[2];   // K = kk+half8 + 16..23
  return __builtin_shufflevector(lo, hi, 0,1,2,3,4,5,6,7,8,9,10,11,12,13,14,15);
}
__device__ __forceinline__ v16bf frag_b_lds(const __bf16* lB, int lane, int wave, int kk) {
  const int col = (wave << 4) + (lane & 15);
  const __bf16* p = lB + (size_t)col * LDB + kk + ((lane >> 4) << 3);
  const v8bf* pv = (const v8bf*)p;
  v8bf lo = pv[0], hi = pv[2];
  return __builtin_shufflevector(lo, hi, 0,1,2,3,4,5,6,7,8,9,10,11,12,13,14,15);
}

// ============ GEMM1: Y = fo_w @ prelu(x) + fo_b.  Block tile: M=32, N=128, K=128 ============
__global__ void __launch_bounds__(256)
fo_gemm_kernel(const float* __restrict__ X, const float* __restrict__ W,
               const float* __restrict__ bias, const float* __restrict__ alphaP,
               float* __restrict__ Y, int M) {
  __shared__ __align__(16) __bf16 sA[32 * LDA];
  __shared__ __align__(16) __bf16 sB[128 * LDB];
  const int tid = threadIdx.x, lane = tid & 31, wave = tid >> 5;
  const int n0b = blockIdx.x * 128, m0 = blockIdx.y * 32;
  stage_w(W, m0, sA, tid);
  stage_x<true>(X, NPIX, n0b, NPIX, alphaP[0], sB, tid);
  __syncthreads();
  v8f acc0 = {}, acc1 = {};
#pragma unroll
  for (int kk = 0; kk < BNCH; kk += 32) {
    v16bf b  = frag_b_lds(sB, lane, wave, kk);
    v16bf a0 = frag_a_lds(sA, lane, kk);
    acc0 = WMMA_BF16(a0, b, acc0);
    v16bf a1 = frag_a_lds(sA + 16 * LDA, lane, kk);
    acc1 = WMMA_BF16(a1, b, acc1);
  }
  const int col = n0b + (wave << 4) + (lane & 15);
  if (col < NPIX) {
    const int h8 = (lane >> 4) * 8;
#pragma unroll
    for (int v = 0; v < 8; ++v) {
      int r0 = m0 + v + h8;
      Y[(size_t)r0 * NPIX + col] = acc0[v] + bias[r0];
      int r1 = r0 + 16;
      Y[(size_t)r1 * NPIX + col] = acc1[v] + bias[r1];
    }
  }
}

// ---- fold: Z[m,t] = sum_{k: 0<=t+100-50k<100} Y[m, t+100-50k, k]
__global__ void fold_kernel(const float* __restrict__ Y, float* __restrict__ Z, int M) {
  const int idx = blockIdx.x * 256 + threadIdx.x;
  if (idx >= M * NFRM) return;
  const int m = idx / NFRM, t = idx % NFRM;
  int kmax = (t + CKSZ) / 50; if (kmax > KFR - 1) kmax = KFR - 1;
  int kmin = t / 50 + 1;
  const float* Ym = Y + (size_t)m * NPIX;
  float s = 0.f;
  for (int k = kmin; k <= kmax; ++k) s += Ym[(t + CKSZ - 50 * k) * KFR + k];
  Z[idx] = s;
}

// ============ gated: MID = tanh(no@Z+bu) * sigmoid(ng@Z+bg). M=128 (4 blocks of 32) ============
__global__ void __launch_bounds__(256)
gated_gemm_kernel(const float* __restrict__ Z, const float* __restrict__ Wu,
                  const float* __restrict__ bu, const float* __restrict__ Wg,
                  const float* __restrict__ bg, float* __restrict__ MID) {
  __shared__ __align__(16) __bf16 sA[64 * LDA];   // Wu rows 0..31, Wg rows 32..63
  __shared__ __align__(16) __bf16 sB[128 * LDB];
  const int tid = threadIdx.x, lane = tid & 31, wave = tid >> 5;
  const int j = blockIdx.z;
  const float* X = Z   + (size_t)j * BNCH * NFRM;
  float*       O = MID + (size_t)j * BNCH * NFRM;
  const int n0b = blockIdx.x * 128, m0 = blockIdx.y * 32;
  stage_w(Wu, m0, sA, tid);
  stage_w(Wg, m0, sA + 32 * LDA, tid);
  stage_x<false>(X, NFRM, n0b, NFRM, 0.f, sB, tid);
  __syncthreads();
  v8f u0 = {}, u1 = {}, g0 = {}, g1 = {};
#pragma unroll
  for (int kk = 0; kk < BNCH; kk += 32) {
    v16bf b = frag_b_lds(sB, lane, wave, kk);
    u0 = WMMA_BF16(frag_a_lds(sA,            lane, kk), b, u0);
    u1 = WMMA_BF16(frag_a_lds(sA + 16 * LDA, lane, kk), b, u1);
    g0 = WMMA_BF16(frag_a_lds(sA + 32 * LDA, lane, kk), b, g0);
    g1 = WMMA_BF16(frag_a_lds(sA + 48 * LDA, lane, kk), b, g1);
  }
  const int col = n0b + (wave << 4) + (lane & 15);
  if (col < NFRM) {
    const int h8 = (lane >> 4) * 8;
#pragma unroll
    for (int v = 0; v < 8; ++v) {
      int r0 = m0 + v + h8, r1 = r0 + 16;
      float uu0 = tanhf(u0[v] + bu[r0]);
      float gg0 = 1.0f / (1.0f + __expf(-(g0[v] + bg[r0])));
      O[(size_t)r0 * NFRM + col] = uu0 * gg0;
      float uu1 = tanhf(u1[v] + bu[r1]);
      float gg1 = 1.0f / (1.0f + __expf(-(g1[v] + bg[r1])));
      O[(size_t)r1 * NFRM + col] = uu1 * gg1;
    }
  }
}

// ============ mask: SRC = relu(mn@MID) * mixture_w[b]. M=64 (2 blocks of 32) ============
__global__ void __launch_bounds__(256)
mask_gemm_kernel(const float* __restrict__ MID, const float* __restrict__ Wm,
                 const float* __restrict__ mwb, float* __restrict__ SRC) {
  __shared__ __align__(16) __bf16 sA[32 * LDA];
  __shared__ __align__(16) __bf16 sB[128 * LDB];
  const int tid = threadIdx.x, lane = tid & 31, wave = tid >> 5;
  const int j = blockIdx.z;
  const float* X = MID + (size_t)j * BNCH  * NFRM;
  float*       O = SRC + (size_t)j * IN_CH * NFRM;
  const int n0b = blockIdx.x * 128, m0 = blockIdx.y * 32;
  stage_w(Wm, m0, sA, tid);
  stage_x<false>(X, NFRM, n0b, NFRM, 0.f, sB, tid);
  __syncthreads();
  v8f acc0 = {}, acc1 = {};
#pragma unroll
  for (int kk = 0; kk < BNCH; kk += 32) {
    v16bf b = frag_b_lds(sB, lane, wave, kk);
    acc0 = WMMA_BF16(frag_a_lds(sA,            lane, kk), b, acc0);
    acc1 = WMMA_BF16(frag_a_lds(sA + 16 * LDA, lane, kk), b, acc1);
  }
  const int col = n0b + (wave << 4) + (lane & 15);
  if (col < NFRM) {
    const int h8 = (lane >> 4) * 8;
#pragma unroll
    for (int v = 0; v < 8; ++v) {
      int r0 = m0 + v + h8, r1 = r0 + 16;
      O[(size_t)r0 * NFRM + col] = fmaxf(acc0[v], 0.f) * mwb[(size_t)r0 * NFRM + col];
      O[(size_t)r1 * NFRM + col] = fmaxf(acc1[v], 0.f) * mwb[(size_t)r1 * NFRM + col];
    }
  }
}

// ---- transposed conv (K=16, stride 8) fused with ground-truth hard select
__global__ void tconv_kernel(const float* __restrict__ SRC, const float* __restrict__ w,
                             const int* __restrict__ gt, float* __restrict__ out,
                             int n, int writeWhenZero) {
  const int j = blockIdx.y;
  const int tau = blockIdx.x * blockDim.x + threadIdx.x;
  if (tau >= TOUT) return;
  const bool z = (gt[n >> 1] == 0);
  if (z != (writeWhenZero != 0)) return;
  const float* sj = SRC + (size_t)j * IN_CH * NFRM;
  int thi = tau >> 3;           if (thi > NFRM - 1) thi = NFRM - 1;
  int tlo = (tau >= 15) ? ((tau - 8) >> 3) : 0;     // ceil((tau-15)/8)
  float acc = 0.f;
  for (int t = tlo; t <= thi; ++t) {
    const int tap = tau - (t << 3);                 // in [0,16)
#pragma unroll
    for (int ic = 0; ic < IN_CH; ++ic)
      acc += sj[(size_t)ic * NFRM + t] * w[ic * 16 + tap];
  }
  out[((size_t)(n * 3 + j)) * TOUT + tau] = acc;
}

// ---- zero pad: decoder-0 has only 2 sources; zero source #2 where gt==0
__global__ void padzero_kernel(const int* __restrict__ gt, float* __restrict__ out) {
  const int idx = blockIdx.x * 256 + threadIdx.x;
  if (idx >= NSAMP * TOUT) return;
  const int ns = idx / TOUT;
  if (gt[ns >> 1] != 0) return;
  out[((size_t)(ns * 3 + 2)) * TOUT + (idx % TOUT)] = 0.f;
}

// ---- selector head
__global__ void selector_kernel(const float* __restrict__ X0, const float* __restrict__ s1w,
                                const float* __restrict__ s1b, const float* __restrict__ s2w,
                                const float* __restrict__ s2b, float* __restrict__ selOut) {
  const int n = blockIdx.x;
  const float* X = X0 + (size_t)n * BNCH * NPIX;
  __shared__ float xbar[BNCH];
  __shared__ float h[IN_CH];
  const int t = threadIdx.x;   // blockDim = 128
  {
    const float4* row4 = (const float4*)(X + (size_t)t * NPIX);
    float s = 0.f;
    for (int i = 0; i < NPIX / 4; ++i) { float4 v = row4[i]; s += v.x + v.y + v.z + v.w; }
    xbar[t] = s * (1.0f / (float)NPIX);
  }
  __syncthreads();
  if (t < IN_CH) {
    float a = s1b[t];
    for (int c = 0; c < BNCH; ++c) a += s1w[t * BNCH + c] * xbar[c];
    h[t] = fmaxf(a, 0.f);
  }
  __syncthreads();
  if (t < 2) {
    float a = s2b[t];
    for (int o = 0; o < IN_CH; ++o) a += s2w[t * IN_CH + o] * h[o];
    selOut[n * 2 + t] = a;
  }
}

extern "C" void kernel_launch(void* const* d_in, const int* in_sizes, int n_in,
                              void* d_out, int out_size, void* d_ws, size_t ws_size,
                              hipStream_t stream) {
  const float* outputs = (const float*)d_in[0];
  const float* mixw    = (const float*)d_in[1];
  const int*   gt      = (const int*)d_in[2];
  const float* A[2]   = { (const float*)d_in[3],  (const float*)d_in[12] };
  const float* FOW[2] = { (const float*)d_in[4],  (const float*)d_in[13] };
  const float* FOB[2] = { (const float*)d_in[5],  (const float*)d_in[14] };
  const float* NOW[2] = { (const float*)d_in[6],  (const float*)d_in[15] };
  const float* NOB[2] = { (const float*)d_in[7],  (const float*)d_in[16] };
  const float* NGW[2] = { (const float*)d_in[8],  (const float*)d_in[17] };
  const float* NGB[2] = { (const float*)d_in[9],  (const float*)d_in[18] };
  const float* MNW[2] = { (const float*)d_in[10], (const float*)d_in[19] };
  const float* TCW[2] = { (const float*)d_in[11], (const float*)d_in[20] };
  const float* s1w = (const float*)d_in[21];
  const float* s1b = (const float*)d_in[22];
  const float* s2w = (const float*)d_in[23];
  const float* s2b = (const float*)d_in[24];
  float* out = (float*)d_out;

  // L2-resident per-(sample,decoder) workspace: 19.5MB + 9.5MB + 4.8MB
  float* Y   = (float*)d_ws;                       // up to 384*12700
  float* Z   = Y + (size_t)384 * NPIX;             // up to 3*128*6200
  float* SRC = Z + (size_t)3 * BNCH * NFRM;        // up to 3*64*6200
  float* MID = Y;                                  // alias: Y dead after fold

  selector_kernel<<<NSAMP, 128, 0, stream>>>(outputs, s1w, s1b, s2w, s2b,
                                             out + (size_t)NSAMP * 3 * TOUT);
  padzero_kernel<<<(NSAMP * TOUT + 255) / 256, 256, 0, stream>>>(gt, out);

  for (int n = 0; n < NSAMP; ++n) {
    const float* Xn  = outputs + (size_t)n * BNCH * NPIX;
    const float* mwb = mixw + (size_t)(n >> 1) * IN_CH * NFRM;
    for (int d = 0; d < 2; ++d) {
      const int nsrc = d ? 3 : 2;
      const int M = nsrc * BNCH;
      dim3 g1((NPIX + 127) / 128, M / 32);
      fo_gemm_kernel<<<g1, 256, 0, stream>>>(Xn, FOW[d], FOB[d], A[d], Y, M);
      fold_kernel<<<(M * NFRM + 255) / 256, 256, 0, stream>>>(Y, Z, M);
      dim3 g2((NFRM + 127) / 128, BNCH / 32, nsrc);
      gated_gemm_kernel<<<g2, 256, 0, stream>>>(Z, NOW[d], NOB[d], NGW[d], NGB[d], MID);
      dim3 g3((NFRM + 127) / 128, IN_CH / 32, nsrc);
      mask_gemm_kernel<<<g3, 256, 0, stream>>>(MID, MNW[d], mwb, SRC);
      dim3 g4((TOUT + 255) / 256, nsrc);
      tconv_kernel<<<g4, 256, 0, stream>>>(SRC, TCW[d], gt, out, n, d == 0 ? 1 : 0);
    }
  }
}
